// AdaptiveJacobianPrunedViT_88673894793323
// MI455X (gfx1250) — matrix operand order repaired and checked by reading, not compiled, not run on hardware
//
#include <hip/hip_runtime.h>
#include <hip/hip_bf16.h>
#include <math.h>

// ---------------- model constants ----------------
#define NB     16      // batch
#define BDIM   768     // D
#define NH     12      // heads
#define HD     64      // head dim
#define NLAYER 12
#define DFF    3072
#define TTOK0  577     // tokens at entry (576 patches + cls)
#define MAXT   592     // ceil16(577)

// ---------------- WMMA plumbing ----------------
typedef __attribute__((ext_vector_type(16))) __bf16 v16bf;
typedef __attribute__((ext_vector_type(8)))  float  v8f;

struct __align__(16) U4 { unsigned x0, x1, x2, x3; };
struct __align__(8)  F2 { float a, b; };

union FragV { v16bf v; U4 q[2]; };   // 16 bf16 = 32B = 2 x b128

__device__ __forceinline__ unsigned short f2bf(float f) {
  union { float f; unsigned u; } c; c.f = f;
  unsigned r = c.u + 0x7FFFu + ((c.u >> 16) & 1u);   // round-to-nearest-even
  return (unsigned short)(r >> 16);
}
__device__ __forceinline__ unsigned pack2bf(float a, float b) {
  return (unsigned)f2bf(a) | ((unsigned)f2bf(b) << 16);
}
__device__ __forceinline__ float bf2f(unsigned short u) {
  union { unsigned u; float f; } c; c.u = ((unsigned)u) << 16; return c.f;
}
__device__ __forceinline__ v8f zero8() {
  v8f z = {0.f,0.f,0.f,0.f,0.f,0.f,0.f,0.f};
  return z;
}
__device__ __forceinline__ v8f wmma_bf16(const FragV& a, const FragV& b, v8f c) {
  return __builtin_amdgcn_wmma_f32_16x16x32_bf16(false, a.v, false, b.v,
                                                 (short)0, c, false, false);
}

// A-fragment (16x32): lane = M row; two contiguous 16B chunks at dword
// offsets {group*4} and {8+group*4} of a K-contiguous packed-bf16 row.
__device__ __forceinline__ void load_afrag_lds(FragV& f, const unsigned* row, int group) {
  f.q[0] = *(const U4*)(row + group * 4);
  f.q[1] = *(const U4*)(row + 8 + group * 4);
}
// B-fragment (32x16): lane = N col; two 16B chunks at element offsets
// {group*8} and {16+group*8} of a K-contiguous bf16 column (LDS or global).
__device__ __forceinline__ void load_bfrag_g(FragV& f, const unsigned short* col, int group) {
  const unsigned short* p = col + group * 8;
  f.q[0] = *(const U4*)(p);
  f.q[1] = *(const U4*)(p + 16);
}

// ---------------- generic GEMM ----------------
// C = gelu?(A @ B^T' + bias) (+res).  A: MxK f32 row-major.
// BT: pre-converted bf16 weights, layout (N,K) (K-contiguous). K % 32 == 0.
// Block tile 128x128, K-chunk 32. 256 thr = 8 waves (4M x 2N), 2x4 WMMA each.
// B-fragments stream straight from global (hits L2: weights reused by all
// M-blocks); only the A tile is staged in LDS (f32 -> packed bf16 pairs).
template<bool BIAS, bool RES, bool GELU>
__global__ __launch_bounds__(256) void gemm_bf16(
    const float* __restrict__ A, const unsigned short* __restrict__ BT,
    const float* __restrict__ bias, const float* __restrict__ res,
    float* __restrict__ C, int M, int N, int K)
{
  // row stride 20 dwords (80B): 16B-aligned rows, 20*r mod 64 -> 16 banks.
  __shared__ __align__(16) unsigned As[128][20];  // [row][kpair]

  const int tid   = threadIdx.x;
  const int lane  = tid & 31, wave = tid >> 5;
  const int wm    = wave >> 1, wn = wave & 1;
  const int group = lane >> 4, ml = lane & 15;
  const int m0 = blockIdx.y * 128, n0 = blockIdx.x * 128;

  // per-wave B column base pointers (clamped: OOB cols feed garbage into
  // accumulators that the epilogue never stores)
  const unsigned short* bbase[4];
  #pragma unroll
  for (int tn = 0; tn < 4; ++tn) {
    int col = n0 + wn * 64 + tn * 16 + ml;
    if (col >= N) col = N - 1;
    bbase[tn] = BT + (size_t)col * K;
  }

  v8f acc[2][4];
  #pragma unroll
  for (int i = 0; i < 2; ++i)
    #pragma unroll
    for (int j = 0; j < 4; ++j) acc[i][j] = zero8();

  for (int k0 = 0; k0 < K; k0 += 32) {
    // stage A 128x32: clamped float2 loads, pack to bf16 pairs
    #pragma unroll
    for (int i = 0; i < 8; ++i) {
      int e = tid + i * 256;            // 128 rows x 16 pairs
      int r = e >> 4, j = e & 15;
      int rr = m0 + r; if (rr >= M) rr = M - 1;       // clamp, no branch
      const float* ap = A + (size_t)rr * K + k0;
      F2 vv = *(const F2*)(ap + 2 * j);
      if (i == 0) __builtin_prefetch(ap + 32, 0, 0);  // next K slab (spec.)
      As[r][j] = pack2bf(vv.a, vv.b);
    }
    __syncthreads();

    FragV af[2], bfq[4];
    #pragma unroll
    for (int tm = 0; tm < 2; ++tm)
      load_afrag_lds(af[tm], &As[wm * 32 + tm * 16 + ml][0], group);
    #pragma unroll
    for (int tn = 0; tn < 4; ++tn)
      load_bfrag_g(bfq[tn], bbase[tn] + k0, group);
    #pragma unroll
    for (int tm = 0; tm < 2; ++tm)
      #pragma unroll
      for (int tn = 0; tn < 4; ++tn)
        acc[tm][tn] = wmma_bf16(af[tm], bfq[tn], acc[tm][tn]);
    __syncthreads();
  }

  // epilogue: C layout lane=N col, vgpr r => M = r + 8*group
  #pragma unroll
  for (int tm = 0; tm < 2; ++tm)
    for (int tn = 0; tn < 4; ++tn)
      #pragma unroll
      for (int r = 0; r < 8; ++r) {
        int row = m0 + wm * 32 + tm * 16 + r + (group << 3);
        int col = n0 + wn * 64 + tn * 16 + ml;
        if (row < M && col < N) {
          float vv = acc[tm][tn][r];
          if (BIAS) vv += bias[col];
          if (GELU) vv = 0.5f * vv * (1.f + erff(vv * 0.70710678118654752f));
          if (RES)  vv += res[(size_t)row * N + col];
          C[(size_t)row * N + col] = vv;
        }
      }
}

// ---------------- LayerNorm over rows of width 768 ----------------
__global__ __launch_bounds__(256) void ln_rows(
    const float* __restrict__ in, float* __restrict__ out,
    const float* __restrict__ g, const float* __restrict__ b)
{
  __shared__ float red[256];
  const int row = blockIdx.x;
  const float* x = in + (size_t)row * BDIM;
  float s = 0.f;
  for (int c = threadIdx.x; c < BDIM; c += 256) s += x[c];
  red[threadIdx.x] = s; __syncthreads();
  for (int o = 128; o > 0; o >>= 1) {
    if (threadIdx.x < o) red[threadIdx.x] += red[threadIdx.x + o];
    __syncthreads();
  }
  float mu = red[0] * (1.f / BDIM);
  __syncthreads();
  float vr = 0.f;
  for (int c = threadIdx.x; c < BDIM; c += 256) { float d = x[c] - mu; vr += d * d; }
  red[threadIdx.x] = vr; __syncthreads();
  for (int o = 128; o > 0; o >>= 1) {
    if (threadIdx.x < o) red[threadIdx.x] += red[threadIdx.x + o];
    __syncthreads();
  }
  float rstd = rsqrtf(red[0] * (1.f / BDIM) + 1e-6f);
  float* y = out + (size_t)row * BDIM;
  for (int c = threadIdx.x; c < BDIM; c += 256)
    y[c] = (x[c] - mu) * rstd * g[c] + b[c];
}

// ---------------- fused attention ----------------
// q,k: (B*H, T, 64) bf16.  vt: (B*H, 64, Tpad) bf16 (transposed, zero-padded).
// One block = one (b,h) x one 16-query tile. Scores live in a 16 x MAXT LDS
// row buffer; probs re-packed to bf16 for the P@V WMMAs.
template<bool OUT>
__global__ __launch_bounds__(256) void attn_kernel(
    const unsigned short* __restrict__ q, const unsigned short* __restrict__ k,
    const unsigned short* __restrict__ vt, float* __restrict__ out,
    float* __restrict__ imp, const float* __restrict__ vnorm,
    int T, int Tpad, float invBH)
{
  __shared__ __align__(16) float          S[16][MAXT];
  __shared__ __align__(16) unsigned short Pb[16][MAXT];   // row = 1184B = 74*16
  __shared__ __align__(16) unsigned       qs[16][36];     // 32 used; stride 36
  __shared__ float red[16][16];
  __shared__ float rowstat[16];

  const int tid = threadIdx.x, lane = tid & 31, wave = tid >> 5;
  const int group = lane >> 4, ml = lane & 15;
  const int bh = blockIdx.x, b = bh / NH, hh = bh % NH;
  const int q0 = blockIdx.y * 16;
  const float scale = 0.125f;   // HD^-0.5

  // stage Q tile 16x64 bf16 (dword copies; clamped rows feed rows never stored)
  for (int e = tid; e < 16 * 32; e += 256) {
    int rr = e >> 5, j = e & 31;
    int qr = q0 + rr; if (qr >= T) qr = T - 1;
    qs[rr][j] = ((const unsigned*)q)[((size_t)bh * T + qr) * 32 + j];
  }
  __syncthreads();

  FragV af[2];
  #pragma unroll
  for (int kc = 0; kc < 2; ++kc)
    load_afrag_lds(af[kc], &qs[ml][0] + kc * 16, group);

  // S = (Q K^T) * scale : key tiles distributed across 8 waves
  for (int jt = wave; jt * 16 < Tpad; jt += 8) {
    v8f acc = zero8();
    int j = jt * 16 + ml;
    int jc = (j < T) ? j : T - 1;                       // clamp, mask below
    const unsigned short* kr = k + ((size_t)bh * T + jc) * HD;
    #pragma unroll
    for (int kc = 0; kc < 2; ++kc) {
      FragV bb;
      load_bfrag_g(bb, kr + kc * 32, group);
      acc = wmma_bf16(af[kc], bb, acc);
    }
    #pragma unroll
    for (int r = 0; r < 8; ++r) {
      int m = r + (group << 3);
      S[m][j] = (j < T) ? acc[r] * scale : -1.0e30f;
    }
  }
  __syncthreads();

  // row-wise softmax (16 lanes per row)
  {
    int r = tid >> 4, c16 = tid & 15;
    float mx = -3.0e30f;
    for (int j = c16; j < Tpad; j += 16) mx = fmaxf(mx, S[r][j]);
    red[r][c16] = mx; __syncthreads();
    if (c16 == 0) {
      float m = red[r][0];
      for (int i = 1; i < 16; ++i) m = fmaxf(m, red[r][i]);
      rowstat[r] = m;
    }
    __syncthreads();
    float rm = rowstat[r];
    float sm = 0.f;
    for (int j = c16; j < Tpad; j += 16) {
      float e = __expf(S[r][j] - rm); S[r][j] = e; sm += e;
    }
    red[r][c16] = sm; __syncthreads();
    if (c16 == 0) {
      float s0 = 0.f;
      for (int i = 0; i < 16; ++i) s0 += red[r][i];
      rowstat[r] = s0;
    }
    __syncthreads();
    float inv = 1.f / rowstat[r];
    for (int j = c16; j < Tpad; j += 16) {
      float p = S[r][j] * inv; S[r][j] = p; Pb[r][j] = f2bf(p);
    }
  }
  __syncthreads();

  if (!OUT) {
    // importance: imp[j] += invBH * (sum_q P[q,j]) * ||v_j||
    for (int j = tid; j < T; j += 256) {
      float cs = 0.f;
      int rmax = T - q0; if (rmax > 16) rmax = 16;
      for (int rr = 0; rr < rmax; ++rr) cs += S[rr][j];
      atomicAdd(&imp[j], cs * vnorm[(size_t)bh * T + j] * invBH);
    }
  } else {
    // O = P @ V : waves 0..3 each own 16 output columns (d)
    if (wave < 4) {
      v8f acc = zero8();
      const int d = wave * 16 + ml;
      const unsigned short* vrow = vt + ((size_t)bh * HD + d) * Tpad;
      for (int kt = 0; kt < Tpad; kt += 32) {
        FragV ap, bp;
        load_afrag_lds(ap, (const unsigned*)&Pb[ml][0] + (kt >> 1), group);
        load_bfrag_g(bp, vrow + kt, group);
        acc = wmma_bf16(ap, bp, acc);
      }
      #pragma unroll
      for (int r = 0; r < 8; ++r) {
        int qr = q0 + r + (group << 3);
        if (qr < T)
          out[((size_t)b * T + qr) * BDIM + hh * HD + d] = acc[r];
      }
    }
  }
}

// ---------------- elementwise / data-movement kernels ----------------
// f32 (L,K,N) weights -> bf16 (L,N,K) (one-time per launch)
__global__ void transw_k(const float* __restrict__ w, unsigned short* __restrict__ wt,
                         int K, int N, int total) {
  int idx = blockIdx.x * 256 + threadIdx.x;
  if (idx >= total) return;
  int kk = idx % K;
  int n  = (idx / K) % N;
  int l  = idx / (K * N);
  wt[idx] = f2bf(w[(size_t)l * K * N + (size_t)kk * N + n]);
}

__global__ void im2col_k(const float* __restrict__ x, float* __restrict__ p, int total) {
  int idx = blockIdx.x * 256 + threadIdx.x;
  if (idx >= total) return;
  int c  = idx % 768;
  int pi = (idx / 768) % 576;
  int b  = idx / (768 * 576);
  int ch = c >> 8, py = (c >> 4) & 15, px = c & 15;
  int gy = pi / 24, gx = pi % 24;
  p[idx] = x[(((size_t)b * 3 + ch) * 384 + gy * 16 + py) * 384 + gx * 16 + px];
}

__global__ void add_cls_pos_k(const float* __restrict__ tok, const float* __restrict__ cls,
                              const float* __restrict__ pos, float* __restrict__ h, int total) {
  int idx = blockIdx.x * 256 + threadIdx.x;
  if (idx >= total) return;
  int c = idx % 768;
  int t = (idx / 768) % TTOK0;
  int b = idx / (768 * TTOK0);
  float vv = (t == 0) ? cls[c] : tok[((size_t)b * 576 + (t - 1)) * 768 + c];
  h[idx] = vv + pos[(size_t)t * 768 + c];
}

// (B,T,3,H,HD) f32 -> q/k/v (B,H,T,HD) bf16
__global__ void permute_qkv_k(const float* __restrict__ src, unsigned short* __restrict__ q,
                              unsigned short* __restrict__ k, unsigned short* __restrict__ v,
                              int T, int total) {
  int idx = blockIdx.x * 256 + threadIdx.x;
  if (idx >= total) return;
  int d  = idx & 63;
  int hh = (idx >> 6) % NH;
  int s  = (idx / 768) % 3;
  int t  = (idx / 2304) % T;
  int b  = idx / (2304 * T);
  unsigned short val = f2bf(src[idx]);
  unsigned short* dst = (s == 0) ? q : (s == 1) ? k : v;
  dst[(((size_t)b * NH + hh) * T + t) * HD + d] = val;
}

__global__ void vnorm_k(const unsigned short* __restrict__ v, float* __restrict__ vn, int total) {
  int idx = blockIdx.x * 256 + threadIdx.x;
  if (idx >= total) return;
  const unsigned short* p = v + (size_t)idx * HD;
  float s = 0.f;
  #pragma unroll
  for (int d = 0; d < HD; ++d) { float f = bf2f(p[d]); s += f * f; }
  vn[idx] = sqrtf(s + 1e-6f);
}

// (B*H, T, HD) bf16 -> (B*H, HD, Tpad) bf16 with zero padding
__global__ void transpose_v_k(const unsigned short* __restrict__ v,
                              unsigned short* __restrict__ vt,
                              int T, int Tpad, int total) {
  int idx = blockIdx.x * 256 + threadIdx.x;
  if (idx >= total) return;
  int t = idx % Tpad;
  int d = (idx / Tpad) % HD;
  int r = idx / (Tpad * HD);
  vt[idx] = (t < T) ? v[((size_t)r * T + t) * HD + d] : (unsigned short)0;
}

// exact top-k (rank select, ties -> lower index; output sorted ascending)
__global__ __launch_bounds__(640) void topk_kernel(const float* __restrict__ imp,
                                                   int* __restrict__ keep,
                                                   int T, int tgt) {
  __shared__ float si[640];
  __shared__ int   fl[640];
  int i = threadIdx.x;
  si[i] = (i < T) ? imp[i] : -3.0e30f;
  fl[i] = 0;
  __syncthreads();
  if (i >= 1 && i < T) {
    float v = si[i];
    int rank = 0;
    for (int j = 1; j < T; ++j) {
      float w = si[j];
      rank += (w > v) || (w == v && j < i);
    }
    fl[i] = (rank < tgt) ? 1 : 0;
  }
  __syncthreads();
  if (i == 0) keep[0] = 0;
  if (i >= 1 && i < T && fl[i]) {
    int pos = 1;
    for (int j = 1; j < i; ++j) pos += fl[j];
    keep[pos] = i;
  }
}

__global__ void gather_rows_f32(const float* __restrict__ src, float* __restrict__ dst,
                                const int* __restrict__ keep, int R, int Tin, int Tout, int C) {
  int idx = blockIdx.x * 256 + threadIdx.x;
  if (idx >= R * Tout * C) return;
  int c = idx % C;
  int t = (idx / C) % Tout;
  int r = idx / (C * Tout);
  dst[idx] = src[((size_t)r * Tin + keep[t]) * C + c];
}

__global__ void gather_rows_u16(const unsigned short* __restrict__ src,
                                unsigned short* __restrict__ dst,
                                const int* __restrict__ keep, int R, int Tin, int Tout, int C) {
  int idx = blockIdx.x * 256 + threadIdx.x;
  if (idx >= R * Tout * C) return;
  int c = idx % C;
  int t = (idx / C) % Tout;
  int r = idx / (C * Tout);
  dst[idx] = src[((size_t)r * Tin + keep[t]) * C + c];
}

__global__ void gather_cls_k(const float* __restrict__ h, float* __restrict__ o,
                             int T, int total) {
  int idx = blockIdx.x * 256 + threadIdx.x;
  if (idx >= total) return;
  int c = idx % 768, b = idx / 768;
  o[idx] = h[(size_t)b * T * 768 + c];
}

// ---------------- host orchestration ----------------
extern "C" void kernel_launch(void* const* d_in, const int* in_sizes, int n_in,
                              void* d_out, int out_size, void* d_ws, size_t ws_size,
                              hipStream_t stream)
{
  (void)in_sizes; (void)n_in; (void)out_size; (void)ws_size;
  const float* x       = (const float*)d_in[0];
  const float* patch_w = (const float*)d_in[1];
  const float* patch_b = (const float*)d_in[2];
  const float* cls_tok = (const float*)d_in[3];
  const float* pos_emb = (const float*)d_in[4];
  const float* ln1_g   = (const float*)d_in[5];
  const float* ln1_b   = (const float*)d_in[6];
  const float* qkv_w   = (const float*)d_in[7];
  const float* qkv_b   = (const float*)d_in[8];
  const float* proj_w  = (const float*)d_in[9];
  const float* proj_b  = (const float*)d_in[10];
  const float* ln2_g   = (const float*)d_in[11];
  const float* ln2_b   = (const float*)d_in[12];
  const float* fc1_w   = (const float*)d_in[13];
  const float* fc1_b   = (const float*)d_in[14];
  const float* fc2_w   = (const float*)d_in[15];
  const float* fc2_b   = (const float*)d_in[16];
  const float* norm_g  = (const float*)d_in[17];
  const float* norm_b  = (const float*)d_in[18];
  const float* head_w  = (const float*)d_in[19];
  const float* head_b  = (const float*)d_in[20];

  // ---- workspace bump allocator ----
  char* wsp = (char*)d_ws;
  size_t off = 0;
  auto alloc = [&](size_t bytes) -> void* {
    void* p = (void*)(wsp + off);
    off += (bytes + 255) & ~(size_t)255;
    return p;
  };
  const size_t SZ_H = (size_t)NB * TTOK0 * BDIM;            // 7.09M elements
  float* hA  = (float*)alloc(SZ_H * 4);
  float* hB  = (float*)alloc(SZ_H * 4);
  float* xn  = (float*)alloc(SZ_H * 4);
  float* ao  = (float*)alloc(SZ_H * 4);
  unsigned short* qb = (unsigned short*)alloc(SZ_H * 2);    // (B,H,T,HD) bf16
  unsigned short* kb = (unsigned short*)alloc(SZ_H * 2);
  unsigned short* vb = (unsigned short*)alloc(SZ_H * 2);
  float* big = (float*)alloc((size_t)NB * TTOK0 * DFF * 4); // 113MB overlay pool
  float* vn  = (float*)alloc((size_t)NB * NH * TTOK0 * 4);
  float* imp = (float*)alloc(MAXT * 4);
  int*  keep = (int*)alloc(MAXT * 4);
  float* cls16  = (float*)alloc((size_t)16 * 768 * 4);
  float* cls16n = (float*)alloc((size_t)16 * 768 * 4);
  // pre-converted bf16 weights, layout (L,N,K)
  unsigned short* patch_wT = (unsigned short*)alloc((size_t)768 * 768 * 2);
  unsigned short* qkv_wT   = (unsigned short*)alloc((size_t)NLAYER * 2304 * 768 * 2);
  unsigned short* proj_wT  = (unsigned short*)alloc((size_t)NLAYER * 768 * 768 * 2);
  unsigned short* fc1_wT   = (unsigned short*)alloc((size_t)NLAYER * 3072 * 768 * 2);
  unsigned short* fc2_wT   = (unsigned short*)alloc((size_t)NLAYER * 768 * 3072 * 2);
  unsigned short* head_wT  = (unsigned short*)alloc((size_t)1000 * 768 * 2);

  // overlays in 'big' (disjoint lifetimes within a layer):
  float* qkvbuf  = big;                                   // (B,T,2304) f32
  float* patches = big;                                   // (B,576,768) f32
  float* mlp     = big;                                   // (B,T,3072) f32
  unsigned short* qg = (unsigned short*)big;              // gathered q (bf16)
  unsigned short* kg = qg + SZ_H;
  unsigned short* vg = kg + SZ_H;
  unsigned short* vt = vg + SZ_H;                         // (B*H, HD, Tpad) bf16

  dim3 thr(256);
  auto grid1 = [](int total) { return dim3((unsigned)((total + 255) / 256)); };

  // ---- one-time weight convert+transpose (f32 (K,N) -> bf16 (N,K)) ----
  { int t0 = 768 * 768;
    transw_k<<<grid1(t0), thr, 0, stream>>>(patch_w, patch_wT, 768, 768, t0); }
  { int t0 = NLAYER * 768 * 2304;
    transw_k<<<grid1(t0), thr, 0, stream>>>(qkv_w, qkv_wT, 768, 2304, t0); }
  { int t0 = NLAYER * 768 * 768;
    transw_k<<<grid1(t0), thr, 0, stream>>>(proj_w, proj_wT, 768, 768, t0); }
  { int t0 = NLAYER * 768 * 3072;
    transw_k<<<grid1(t0), thr, 0, stream>>>(fc1_w, fc1_wT, 768, 3072, t0); }
  { int t0 = NLAYER * 3072 * 768;
    transw_k<<<grid1(t0), thr, 0, stream>>>(fc2_w, fc2_wT, 3072, 768, t0); }
  { int t0 = 768 * 1000;
    transw_k<<<grid1(t0), thr, 0, stream>>>(head_w, head_wT, 768, 1000, t0); }

  // ---- patch embed ----
  { int total = NB * 576 * 768;
    im2col_k<<<grid1(total), thr, 0, stream>>>(x, patches, total); }
  { dim3 g((768 + 127) / 128, (NB * 576 + 127) / 128);
    gemm_bf16<true, false, false><<<g, thr, 0, stream>>>(
        patches, patch_wT, patch_b, nullptr, xn, NB * 576, 768, 768); }
  { int total = NB * TTOK0 * 768;
    add_cls_pos_k<<<grid1(total), thr, 0, stream>>>(xn, cls_tok, pos_emb, hA, total); }

  float* h = hA; float* hAlt = hB;
  int T = TTOK0;
  const float invBH = 1.f / (NB * NH);

  for (int l = 0; l < NLAYER; ++l) {
    // LN1
    ln_rows<<<NB * T, thr, 0, stream>>>(h, xn, ln1_g + l * BDIM, ln1_b + l * BDIM);
    // QKV GEMM -> (B,T,2304) f32, then split to bf16 (B,H,T,HD)
    { dim3 g((2304 + 127) / 128, (NB * T + 127) / 128);
      gemm_bf16<true, false, false><<<g, thr, 0, stream>>>(
          xn, qkv_wT + (size_t)l * 2304 * 768, qkv_b + l * 2304, nullptr,
          qkvbuf, NB * T, 2304, BDIM); }
    { int total = NB * T * 2304;
      permute_qkv_k<<<grid1(total), thr, 0, stream>>>(qkvbuf, qb, kb, vb, T, total); }
    { int total = NB * NH * T;
      vnorm_k<<<grid1(total), thr, 0, stream>>>(vb, vn, total); }
    hipMemsetAsync(imp, 0, (size_t)T * sizeof(float), stream);

    // importance pass (pre-prune attention, column sums only)
    int Tpad = (T + 15) & ~15;
    { dim3 g(NB * NH, Tpad / 16);
      attn_kernel<false><<<g, thr, 0, stream>>>(qb, kb, nullptr, nullptr, imp, vn,
                                                T, Tpad, invBH); }
    // deterministic token schedule (matches reference _target_tokens)
    double frac = (double)(l + 1) / NLAYER;
    int tgt = (int)(576.0 * (1.0 - 0.6 * frac * sqrt(frac)));
    if (tgt < 16) tgt = 16;
    topk_kernel<<<1, 640, 0, stream>>>(imp, keep, T, tgt);
    int Tn = tgt + 1;
    { int total = NB * Tn * BDIM;
      gather_rows_f32<<<grid1(total), thr, 0, stream>>>(h, hAlt, keep, NB, T, Tn, BDIM); }
    { int total = NB * NH * Tn * HD;
      gather_rows_u16<<<grid1(total), thr, 0, stream>>>(qb, qg, keep, NB * NH, T, Tn, HD);
      gather_rows_u16<<<grid1(total), thr, 0, stream>>>(kb, kg, keep, NB * NH, T, Tn, HD);
      gather_rows_u16<<<grid1(total), thr, 0, stream>>>(vb, vg, keep, NB * NH, T, Tn, HD); }
    { float* tmp = h; h = hAlt; hAlt = tmp; }
    T = Tn;
    Tpad = (T + 15) & ~15;
    // transposed, zero-padded V for the P@V fragments
    { int total = NB * NH * HD * Tpad;
      transpose_v_k<<<grid1(total), thr, 0, stream>>>(vg, vt, T, Tpad, total); }

    // attention output pass (post-prune)
    { dim3 g(NB * NH, Tpad / 16);
      attn_kernel<true><<<g, thr, 0, stream>>>(qg, kg, vt, ao, nullptr, nullptr,
                                               T, Tpad, invBH); }
    // proj + residual (in-place on h: each element RMW by one thread)
    { dim3 g((BDIM + 127) / 128, (NB * T + 127) / 128);
      gemm_bf16<true, true, false><<<g, thr, 0, stream>>>(
          ao, proj_wT + (size_t)l * 768 * 768, proj_b + l * BDIM, h, h,
          NB * T, BDIM, BDIM); }
    // LN2
    ln_rows<<<NB * T, thr, 0, stream>>>(h, xn, ln2_g + l * BDIM, ln2_b + l * BDIM);
    // FC1 + exact GELU
    { dim3 g((DFF + 127) / 128, (NB * T + 127) / 128);
      gemm_bf16<true, false, true><<<g, thr, 0, stream>>>(
          xn, fc1_wT + (size_t)l * 3072 * 768, fc1_b + l * DFF, nullptr,
          mlp, NB * T, DFF, BDIM); }
    // FC2 + residual
    { dim3 g((BDIM + 127) / 128, (NB * T + 127) / 128);
      gemm_bf16<true, true, false><<<g, thr, 0, stream>>>(
          mlp, fc2_wT + (size_t)l * 768 * 3072, fc2_b + l * BDIM, h, h,
          NB * T, BDIM, DFF); }
  }

  // ---- head: final LN on cls token + classifier GEMM ----
  { int total = 16 * 768;
    gather_cls_k<<<grid1(total), thr, 0, stream>>>(h, cls16, T, total); }
  ln_rows<<<16, thr, 0, stream>>>(cls16, cls16n, norm_g, norm_b);
  { dim3 g((1000 + 127) / 128, 1);
    gemm_bf16<true, false, false><<<g, thr, 0, stream>>>(
        cls16n, head_wT, head_b, nullptr, (float*)d_out, 16, 1000, 768); }
}
